// HCTargetAwareAttnNP_91311004713265
// MI455X (gfx1250) — matrix operand order repaired and checked by reading, not compiled, not run on hardware
//
#include <hip/hip_runtime.h>
#include <hip/hip_bf16.h>

// ---------------------------------------------------------------------------
// HCTargetAwareAttnNP fused implementation for gfx1250 (MI455X).
// B=4, Nt=128, Nc=256, D=256, d_phi=16, hid=128, H=8, dk=32
// ---------------------------------------------------------------------------

typedef _Float16 v16h __attribute__((ext_vector_type(16)));
typedef _Float16 v8h  __attribute__((ext_vector_type(8)));
typedef float    v8f  __attribute__((ext_vector_type(8)));
typedef unsigned int u32x4 __attribute__((ext_vector_type(4)));
typedef int          i32x4 __attribute__((ext_vector_type(4)));
typedef int          i32x8 __attribute__((ext_vector_type(8)));

#define WMMA_F16(a, b, c) \
  __builtin_amdgcn_wmma_f32_16x16x32_f16(false, (a), false, (b), (short)0, (c), false, false)

// A-matrix fragment (16x32 f16, row-major source, lane holds row = lane%16).
// element e -> K = e + 16*hi (contiguous 16 halfs).
__device__ __forceinline__ v16h frag_a(const _Float16* base, int strideH, int row,
                                       int kk, int hi) {
  return *(const v16h*)(base + row * strideH + kk + 16 * hi);
}

// B-matrix fragment (32x16 f16). Source is W transposed: Wt[n][k] row-major,
// lane holds column n = lane%16. element e -> K = e + 16*hi (contiguous 16 halfs).
__device__ __forceinline__ v16h frag_b(const _Float16* base, int strideH, int row,
                                       int kk, int hi) {
  return *(const v16h*)(base + row * strideH + kk + 16 * hi);
}

// Tensor Data Mover: async DMA of a 16-row x 128-col fp32 tile (row stride 128)
// from global memory into LDS at byte offset `lds_addr`.
// D# layout per CDNA5 ISA (2D, groups 2/3 unused).
__device__ __forceinline__ void tdm_load_tile_16x128_f32(unsigned lds_addr,
                                                         const void* gptr) {
  unsigned long long ga = (unsigned long long)(uintptr_t)gptr;
  u32x4 g0 = { 1u,                                   // count=1, is_restore=0
               lds_addr,                             // lds_addr [63:32]
               (unsigned)(ga & 0xFFFFFFFFu),         // global_addr [95:64]
               (unsigned)((ga >> 32) & 0x1FFFFFFu)   // global_addr [120:96]
                 | (2u << 30) };                     // type=2 ("image")
  i32x8 g1 = { (int)(2u << 16),     // workgroup_mask=0, data_size=2 (4B)
               (int)(128u << 16),   // tensor_dim0[15:0]=128 (bits 63:48)
               (int)(16u << 16),    // tensor_dim0 hi=0, tensor_dim1[15:0]=16
               (int)(128u << 16),   // tensor_dim1 hi=0, tile_dim0=128
               (int)16,             // tile_dim1=16, tile_dim2=0
               (int)128,            // tensor_dim0_stride[31:0]=128
               0, 0 };
  i32x4 z4 = {0, 0, 0, 0};
#if defined(__clang_major__) && __clang_major__ >= 23
  i32x8 z8 = {0, 0, 0, 0, 0, 0, 0, 0};
  __builtin_amdgcn_tensor_load_to_lds(g0, g1, z4, z4, z8, 0);
#else
  __builtin_amdgcn_tensor_load_to_lds(g0, g1, z4, z4, 0);
#endif
}

// ---------------------------------------------------------------------------
// Generic small fp32 GEMM: Y[R,N] = X[R,Kd] @ W[Kd,N] (+ bias). One thread/elem.
// ---------------------------------------------------------------------------
__global__ void proj_kernel(const float* __restrict__ X, const float* __restrict__ W,
                            const float* __restrict__ bias, float* __restrict__ Y,
                            int R, int Kd, int N) {
  int idx = blockIdx.x * blockDim.x + threadIdx.x;
  if (idx >= R * N) return;
  int r = idx / N, n = idx % N;
  float acc = bias ? bias[n] : 0.0f;
  const float* xr = X + (size_t)r * Kd;
  for (int k = 0; k < Kd; ++k) acc = fmaf(xr[k], W[(size_t)k * N + n], acc);
  Y[idx] = acc;
}

// Convert fp32 W[K,N] -> f16 Wt[N,K] (transposed, N-major for WMMA B-frags).
__global__ void convtrans_kernel(const float* __restrict__ src,
                                 _Float16* __restrict__ dst, int K, int N) {
  int idx = blockIdx.x * blockDim.x + threadIdx.x;
  if (idx >= K * N) return;
  int k = idx / N, n = idx % N;
  dst[(size_t)n * K + k] = (_Float16)src[(size_t)k * N + n];
}

// ---------------------------------------------------------------------------
// Fused pairwise K/V + gate + attention kernel.
// Grid: B*Nt blocks, 128 threads (4 wave32).
// ---------------------------------------------------------------------------
__global__ __launch_bounds__(128, 1)
void fused_attn_kernel(
    const float* __restrict__ Q,    // (B*Nt, 256) = R_t@Wq + b
    const float* __restrict__ KT,   // (B*Nt, 256) = R_t@ktgt
    const float* __restrict__ VT,   // (B*Nt, 256) = R_t@vtgt
    const float* __restrict__ KC,   // (B*Nc, 256) = R_ctx@kctx
    const float* __restrict__ VC,   // (B*Nc, 256) = R_ctx@vctx
    const float* __restrict__ At,   // (B*Nt, 128) = phi_t@kphi_w1
    const float* __restrict__ Bt,   // (B*Nt, 128) = phi_t@vphi_w1
    const float* __restrict__ Ac,   // (B*Nc, 128) = phi_c@kphi_w1
    const float* __restrict__ Bc,   // (B*Nc, 128) = phi_c@vphi_w1
    const float* __restrict__ kphi_b1, const float* __restrict__ vphi_b1,
    const float* __restrict__ kphi_b2, const float* __restrict__ vphi_b2,
    const float* __restrict__ gate_b,
    const _Float16* __restrict__ kw2t,  // (256,128) f16: kphi_w2^T
    const _Float16* __restrict__ vw2t,  // (256,128) f16: vphi_w2^T
    const _Float16* __restrict__ gwt,   // (256,768) f16: gate_w^T
    float* __restrict__ ctx)            // (B*Nt, 256)
{
  constexpr int D = 256, HID = 128, NC = 256, H = 8;
  const int trow = blockIdx.x;          // b*Nt + t
  const int b    = trow >> 7;           // Nt = 128
  const int cbase = b * NC;

  const int tid  = threadIdx.x;
  const int wave = tid >> 5;
  const int lane = tid & 31;
  const int ln   = lane & 15;
  const int hi   = lane >> 4;

  __shared__ __align__(32) _Float16 sHK[16 * HID];    // 4 KB relu hidden (K)
  __shared__ __align__(32) _Float16 sHV[16 * HID];    // 4 KB relu hidden (V)
  __shared__ __align__(32) _Float16 sGin[16 * 768];   // 24 KB [K|V||K-V|] -> [Kg|Vg|.]
  __shared__ __align__(32) _Float16 sGAcc[16 * 256];  // 8 KB gate pre-activations
  __shared__ __align__(16) float sAcS[16 * HID];      // 8 KB TDM staging (Ac tile)
  __shared__ __align__(16) float sBcS[16 * HID];      // 8 KB TDM staging (Bc tile)
  __shared__ float sQ[D];
  __shared__ float sAt[HID], sBt[HID];
  __shared__ float sScore[H * 16];

  for (int j = tid; j < D; j += 128) sQ[j] = Q[(size_t)trow * D + j];
  for (int j = tid; j < HID; j += 128) {
    sAt[j] = At[(size_t)trow * HID + j];
    sBt[j] = Bt[(size_t)trow * HID + j];
  }
  __syncthreads();

  // Online-softmax state: thread (h = tid>>4, u = tid&15) owns dims u, u+16 of head h.
  const int h = tid >> 4;
  const int u = tid & 15;
  float m_run = -INFINITY, l_run = 0.0f, acc0 = 0.0f, acc1 = 0.0f;

  for (int ci = 0; ci < NC / 16; ++ci) {
    const int c0 = ci * 16;

    // ---- P0: TDM async DMA of this tile's Ac/Bc rows into LDS ------------
    if (wave == 0) {
      unsigned ldsA = (unsigned)(uintptr_t)(void*)sAcS;
      unsigned ldsB = (unsigned)(uintptr_t)(void*)sBcS;
      tdm_load_tile_16x128_f32(ldsA, (const void*)(Ac + (size_t)(cbase + c0) * HID));
      tdm_load_tile_16x128_f32(ldsB, (const void*)(Bc + (size_t)(cbase + c0) * HID));
      __builtin_amdgcn_s_wait_tensorcnt(0);
    }
    __syncthreads();

    // ---- P1: hidden = relu(At - Ac + b1) in f16 --------------------------
    for (int idx = tid; idx < 16 * HID; idx += 128) {
      int j = idx & 127;
      float hk = sAt[j] - sAcS[idx] + kphi_b1[j];
      float hv = sBt[j] - sBcS[idx] + vphi_b1[j];
      sHK[idx] = (_Float16)fmaxf(hk, 0.0f);
      sHV[idx] = (_Float16)fmaxf(hv, 0.0f);
    }
    __syncthreads();

    // ---- P2: Kphi/Vphi WMMA (16x256, K=128) + broadcast adds -> sGin -----
    for (int q = 0; q < 4; ++q) {
      const int n0 = (wave * 4 + q) * 16;
      const int n  = n0 + ln;
      v8f accK = {0, 0, 0, 0, 0, 0, 0, 0};
      v8f accV = {0, 0, 0, 0, 0, 0, 0, 0};
#pragma unroll 2
      for (int kk = 0; kk < HID; kk += 32) {
        v16h aK = frag_a(sHK, HID, ln, kk, hi);
        v16h bK = frag_b(kw2t, HID, n, kk, hi);
        accK = WMMA_F16(aK, bK, accK);
        v16h aV = frag_a(sHV, HID, ln, kk, hi);
        v16h bV = frag_b(vw2t, HID, n, kk, hi);
        accV = WMMA_F16(aV, bV, accV);
      }
      const float ktn = KT[(size_t)trow * D + n] + kphi_b2[n];
      const float vtn = VT[(size_t)trow * D + n] + vphi_b2[n];
#pragma unroll
      for (int r = 0; r < 8; ++r) {
        int m = r + 8 * hi;   // ctx row within tile
        float Kv = accK[r] + ktn + KC[(size_t)(cbase + c0 + m) * D + n];
        float Vv = accV[r] + vtn + VC[(size_t)(cbase + c0 + m) * D + n];
        sGin[m * 768 + n]       = (_Float16)Kv;
        sGin[m * 768 + 256 + n] = (_Float16)Vv;
        sGin[m * 768 + 512 + n] = (_Float16)fabsf(Kv - Vv);
      }
    }
    __syncthreads();

    // ---- P3a: gate GEMM (16x256, K=768) -> sGAcc ------------------------
    for (int q = 0; q < 4; ++q) {
      const int n0 = (wave * 4 + q) * 16;
      const int n  = n0 + ln;
      __builtin_prefetch(gwt + (size_t)n * 768, 0, 0);
      v8f acc = {0, 0, 0, 0, 0, 0, 0, 0};
#pragma unroll 2
      for (int kk = 0; kk < 768; kk += 32) {
        v16h a = frag_a(sGin, 768, ln, kk, hi);
        v16h bb = frag_b(gwt, 768, n, kk, hi);
        acc = WMMA_F16(a, bb, acc);
      }
#pragma unroll
      for (int r = 0; r < 8; ++r) sGAcc[(r + 8 * hi) * 256 + n] = (_Float16)acc[r];
    }
    __syncthreads();

    // ---- P3b: sigmoid gate, Kg/Vg in place ------------------------------
    for (int q = 0; q < 4; ++q) {
      const int n = (wave * 4 + q) * 16 + ln;
      const float gb = gate_b[n];
#pragma unroll
      for (int r = 0; r < 8; ++r) {
        int m = r + 8 * hi;
        float g  = 1.0f / (1.0f + __expf(-((float)sGAcc[m * 256 + n] + gb)));
        float Kv = (float)sGin[m * 768 + n];
        float Vv = (float)sGin[m * 768 + 256 + n];
        sGin[m * 768 + n]       = (_Float16)(Kv * g);
        sGin[m * 768 + 256 + n] = (_Float16)(Vv * g);
      }
    }
    __syncthreads();

    // ---- P4: scores[h][c] = (Q_h . Kg[c]_h) / sqrt(dk) -------------------
    {
      const int c = tid & 15;          // thread = (head h, ctx row c)
      float s = 0.0f;
#pragma unroll
      for (int d = 0; d < 32; ++d)
        s = fmaf(sQ[h * 32 + d], (float)sGin[c * 768 + h * 32 + d], s);
      sScore[h * 16 + c] = s * 0.17677669529663687f;  // 1/sqrt(32)
    }
    __syncthreads();

    // ---- P5: online softmax accumulate over Vg ---------------------------
    {
      float tmax = -INFINITY;
#pragma unroll
      for (int c = 0; c < 16; ++c) tmax = fmaxf(tmax, sScore[h * 16 + c]);
      float mnew  = fmaxf(m_run, tmax);
      float scale = __expf(m_run - mnew);
      l_run *= scale; acc0 *= scale; acc1 *= scale;
#pragma unroll
      for (int c = 0; c < 16; ++c) {
        float p = __expf(sScore[h * 16 + c] - mnew);
        l_run += p;
        acc0 = fmaf(p, (float)sGin[c * 768 + 256 + h * 32 + u],      acc0);
        acc1 = fmaf(p, (float)sGin[c * 768 + 256 + h * 32 + u + 16], acc1);
      }
      m_run = mnew;
    }
    __syncthreads();
  }

  float inv_l = 1.0f / l_run;
  ctx[(size_t)trow * D + h * 32 + u]      = acc0 * inv_l;
  ctx[(size_t)trow * D + h * 32 + u + 16] = acc1 * inv_l;
}

// ---------------------------------------------------------------------------
extern "C" void kernel_launch(void* const* d_in, const int* in_sizes, int n_in,
                              void* d_out, int out_size, void* d_ws, size_t ws_size,
                              hipStream_t stream) {
  constexpr int B = 4, Nt = 128, Nc = 256, D = 256, DPHI = 16, HID = 128;
  const int RT = B * Nt;   // 512 target rows
  const int RC = B * Nc;   // 1024 context rows

  const float* R_t     = (const float*)d_in[0];
  const float* R_ctx   = (const float*)d_in[1];
  const float* phi_t   = (const float*)d_in[2];
  const float* phi_c   = (const float*)d_in[3];
  const float* Wq_w    = (const float*)d_in[4];
  const float* Wq_b    = (const float*)d_in[5];
  const float* kctx_w  = (const float*)d_in[6];
  const float* ktgt_w  = (const float*)d_in[7];
  const float* kphi_w1 = (const float*)d_in[8];
  const float* kphi_b1 = (const float*)d_in[9];
  const float* kphi_w2 = (const float*)d_in[10];
  const float* kphi_b2 = (const float*)d_in[11];
  const float* vctx_w  = (const float*)d_in[12];
  const float* vtgt_w  = (const float*)d_in[13];
  const float* vphi_w1 = (const float*)d_in[14];
  const float* vphi_b1 = (const float*)d_in[15];
  const float* vphi_w2 = (const float*)d_in[16];
  const float* vphi_b2 = (const float*)d_in[17];
  const float* gate_w  = (const float*)d_in[18];
  const float* gate_b  = (const float*)d_in[19];
  const float* out_w   = (const float*)d_in[20];
  const float* out_b   = (const float*)d_in[21];
  float* out = (float*)d_out;

  // Workspace layout (bytes; all offsets multiples of 256).
  char* w = (char*)d_ws;
  float* Q    = (float*)(w + 0);        // 512*256*4  = 524288
  float* KT   = (float*)(w + 524288);   // 524288
  float* VT   = (float*)(w + 1048576);  // 524288
  float* KC   = (float*)(w + 1572864);  // 1024*256*4 = 1048576
  float* VC   = (float*)(w + 2621440);  // 1048576
  float* At   = (float*)(w + 3670016);  // 512*128*4  = 262144
  float* Btv  = (float*)(w + 3932160);  // 262144
  float* Ac   = (float*)(w + 4194304);  // 1024*128*4 = 524288
  float* Bc   = (float*)(w + 4718592);  // 524288
  float* ctx  = (float*)(w + 5242880);  // 524288
  _Float16* kw2t = (_Float16*)(w + 5767168); // 256*128*2 = 65536
  _Float16* vw2t = (_Float16*)(w + 5832704); // 65536
  _Float16* gwt  = (_Float16*)(w + 5898240); // 256*768*2 = 393216
  (void)ws_size; (void)in_sizes; (void)n_in; (void)out_size;

  auto proj = [&](const float* X, const float* W2, const float* bias, float* Y,
                  int R, int Kd, int N) {
    int total = R * N;
    proj_kernel<<<(total + 255) / 256, 256, 0, stream>>>(X, W2, bias, Y, R, Kd, N);
  };

  // Per-token projections (phi MLP layer-1 decomposed across t/c).
  proj(R_t,   Wq_w,    Wq_b,    Q,   RT, D, D);
  proj(R_t,   ktgt_w,  nullptr, KT,  RT, D, D);
  proj(R_t,   vtgt_w,  nullptr, VT,  RT, D, D);
  proj(R_ctx, kctx_w,  nullptr, KC,  RC, D, D);
  proj(R_ctx, vctx_w,  nullptr, VC,  RC, D, D);
  proj(phi_t, kphi_w1, nullptr, At,  RT, DPHI, HID);
  proj(phi_t, vphi_w1, nullptr, Btv, RT, DPHI, HID);
  proj(phi_c, kphi_w1, nullptr, Ac,  RC, DPHI, HID);
  proj(phi_c, vphi_w1, nullptr, Bc,  RC, DPHI, HID);

  // f16 transposed weights for WMMA B-fragments.
  convtrans_kernel<<<(HID * D + 255) / 256, 256, 0, stream>>>(kphi_w2, kw2t, HID, D);
  convtrans_kernel<<<(HID * D + 255) / 256, 256, 0, stream>>>(vphi_w2, vw2t, HID, D);
  convtrans_kernel<<<(3 * D * D + 255) / 256, 256, 0, stream>>>(gate_w, gwt, 3 * D, D);

  // Fused pairwise K/V + gate + attention (WMMA heavy, TDM-staged phi tiles).
  fused_attn_kernel<<<RT, 128, 0, stream>>>(
      Q, KT, VT, KC, VC, At, Btv, Ac, Bc,
      kphi_b1, vphi_b1, kphi_b2, vphi_b2, gate_b,
      kw2t, vw2t, gwt, ctx);

  // Output projection.
  proj(ctx, out_w, out_b, out, RT, D, D);
}
